// Model_64372969832722
// MI455X (gfx1250) — matrix operand order repaired and checked by reading, not compiled, not run on hardware
//
#include <hip/hip_runtime.h>
#include <hip/hip_bf16.h>

// ---------------------------------------------------------------------------
// 2-layer GAT for MI455X (gfx1250).
//  - GEMMs: v_wmma_f32_16x16x32_bf16, B slab staged in LDS via
//    global_load_async_to_lds_b128 (ASYNCcnt) and read back with ds_load_b128.
//  - Edge phase: float4 gathers + f32 global atomics (L2-resident).
// ---------------------------------------------------------------------------

typedef __attribute__((ext_vector_type(16))) __bf16       v16bf;
typedef __attribute__((ext_vector_type(8)))  float        v8f;
typedef __attribute__((ext_vector_type(4)))  unsigned int v4u;

#define N_NODES 50000
#define DIN     256
#define HEADS   4
#define SLOPE   0.2f

// ---- helpers --------------------------------------------------------------

__device__ __forceinline__ __bf16 f2bf(float f) {
  unsigned u = __float_as_uint(f);
  u = u + 0x7FFFu + ((u >> 16) & 1u);        // round-to-nearest-even
  unsigned short s = (unsigned short)(u >> 16);
  __bf16 r;
  __builtin_memcpy(&r, &s, 2);
  return r;
}

__device__ __forceinline__ void atomicMaxF(float* addr, float val) {
  // classic monotone-int-encoding float max; works with -inf init
  if (val >= 0.0f) atomicMax((int*)addr, __float_as_int(val));
  else             atomicMin((unsigned int*)addr, __float_as_uint(val));
}

// CDNA5 async global->LDS copy (per-lane 16B), tracked by ASYNCcnt.
__device__ __forceinline__ void async_copy_b128(unsigned lds_off,
                                                unsigned long long gaddr) {
  asm volatile("global_load_async_to_lds_b128 %0, %1, off"
               :: "v"(lds_off), "v"(gaddr) : "memory");
}
__device__ __forceinline__ void wait_asynccnt0() {
  asm volatile("s_wait_asynccnt 0x0" ::: "memory");
}

// ---- generic fill / convert ----------------------------------------------

__global__ void fill_f32(float* __restrict__ p, float v, long long n) {
  long long i = (long long)blockIdx.x * blockDim.x + threadIdx.x;
  if (i < n) p[i] = v;
}

__global__ void cvt_f32_bf16(const float* __restrict__ in,
                             __bf16* __restrict__ out, long long n) {
  long long i = (long long)blockIdx.x * blockDim.x + threadIdx.x;
  if (i < n) out[i] = f2bf(in[i]);
}

// ---- weight repack: row-major f32 [Kin x Kout]  ->  WMMA-B bf16 fragments -
// Fragment storage: for tile (nt, ks): 32 lanes x 16 bf16 (32B/lane).
// lane<16 : N = nt*16+lane,    elements j=0..15 -> K = ks*32 + j
// lane>=16: N = nt*16+lane-16, elements j=0..15 -> K = ks*32 + 16 + j

__global__ void pack_w_bf16(const float* __restrict__ W,
                            __bf16* __restrict__ out, int Kin, int Kout) {
  int idx = blockIdx.x * blockDim.x + threadIdx.x;
  int total = Kin * Kout;
  if (idx >= total) return;
  int KS   = Kin >> 5;
  int j    = idx & 15;
  int lane = (idx >> 4) & 31;
  int blk  = idx >> 9;          // nt*KS + ks
  int ks   = blk % KS;
  int nt   = blk / KS;
  int n    = nt * 16 + (lane & 15);
  int k    = ks * 32 + (lane >> 4) * 16 + j;
  out[idx] = f2bf(W[(size_t)k * Kout + n]);
}

// ---- bf16 WMMA GEMM: C[M x Kout] = A[M x Kin] * B  (fp32 accumulate) ------
// Block = 8 waves: 8 m-tiles x one 64-column group.  The packed-B slab for
// the column group (32 KB, contiguous in fragment order) is staged into LDS
// once with async b128 copies and reused by every wave across all K-steps.

__global__ void __launch_bounds__(256)
gemm_bf16_wmma(const __bf16* __restrict__ A, const __bf16* __restrict__ Bp,
               float* __restrict__ C, int M, int Kin, int Kout) {
  __shared__ __attribute__((aligned(16))) __bf16 bs[16384];   // 32 KB slab
  const int KS      = Kin >> 5;                 // K-steps of 32 (= 8)
  const int ngroups = Kout >> 6;                // 64-col groups
  const int mg   = blockIdx.x / ngroups;        // group of 8 m-tiles
  const int ng   = blockIdx.x % ngroups;
  const int lane = threadIdx.x & 31;
  const int wave = threadIdx.x >> 5;

  // ---- stage packed-B slab (4 n-tiles x KS k-steps) into LDS ----
  {
    const size_t slab_bytes = (size_t)4 * KS * 32 * 32;      // 32768
    const char* gsrc = (const char*)(Bp) + (size_t)ng * slab_bytes;
    const unsigned lds0 = (unsigned)(size_t)(void*)bs;       // LDS byte offset
    const int chunks = (int)(slab_bytes / 16 / 256);         // b128 per thread
    for (int i = 0; i < chunks; ++i) {
      unsigned off = ((unsigned)i * 256u + threadIdx.x) * 16u;
      async_copy_b128(lds0 + off, (unsigned long long)(gsrc + off));
    }
    wait_asynccnt0();
  }
  __syncthreads();

  const int mt = mg * 8 + wave;
  if (mt * 16 < M) {                            // wave-uniform guard
    const int hf  = lane >> 4;
    const int l15 = lane & 15;
    // A fragment base: row = mt*16 + l15; contiguous K layout (matches packed B):
    //   elements 0..15 -> K = ks*32 + hf*16 .. +15
    const __bf16* arow = A + (size_t)(mt * 16 + l15) * Kin + hf * 16;

    v8f acc[4];
    acc[0] = v8f{}; acc[1] = v8f{}; acc[2] = v8f{}; acc[3] = v8f{};

    for (int ks = 0; ks < KS; ++ks) {
      union { v4u u[2]; v16bf v; } af;
      const v4u* ap = (const v4u*)(arow + ks * 32);
      af.u[0] = ap[0];                                   // K +0..7
      af.u[1] = ap[1];                                   // K +8..15
      if (ks + 1 < KS)
        __builtin_prefetch(arow + (ks + 1) * 32, 0, 3);  // global_prefetch_b8

#pragma unroll
      for (int t = 0; t < 4; ++t) {
        const v4u* bp = (const v4u*)(bs + ((size_t)(t * KS + ks) * 32 + lane) * 16);
        union { v4u u[2]; v16bf v; } bf;
        bf.u[0] = bp[0];                                 // ds_load_b128
        bf.u[1] = bp[1];
        acc[t] = __builtin_amdgcn_wmma_f32_16x16x32_bf16(
            false, af.v, false, bf.v, (short)0, acc[t], false, false);
      }
    }

    // C layout: VGPR v -> row = mt*16 + v + 8*hf, col = nt*16 + l15
#pragma unroll
    for (int t = 0; t < 4; ++t) {
      const int nt = ng * 4 + t;
      float* cp = C + (size_t)(mt * 16 + 8 * hf) * Kout + nt * 16 + l15;
#pragma unroll
      for (int v = 0; v < 8; ++v) cp[(size_t)v * Kout] = acc[t][v];
    }
  }
}

// ---- attention logits: el[n,h] = <fs[n,h,:], al[h,:]> ---------------------

__global__ void el_er_kernel(const float* __restrict__ fs,
                             const float* __restrict__ fd,
                             const float* __restrict__ al,
                             const float* __restrict__ ar,
                             float* __restrict__ el, float* __restrict__ er,
                             int n_nodes, int D) {
  int idx = blockIdx.x * blockDim.x + threadIdx.x;
  if (idx >= n_nodes * HEADS) return;
  int n = idx / HEADS, h = idx % HEADS;
  const float* f1 = fs + (size_t)n * HEADS * D + h * D;
  const float* f2 = fd + (size_t)n * HEADS * D + h * D;
  const float* a1 = al + h * D;
  const float* a2 = ar + h * D;
  float s1 = 0.f, s2 = 0.f;
  for (int d = 0; d < D; d += 4) {
    float4 x1 = *(const float4*)(f1 + d), w1 = *(const float4*)(a1 + d);
    float4 x2 = *(const float4*)(f2 + d), w2 = *(const float4*)(a2 + d);
    s1 += x1.x * w1.x + x1.y * w1.y + x1.z * w1.z + x1.w * w1.w;
    s2 += x2.x * w2.x + x2.y * w2.y + x2.z * w2.z + x2.w * w2.w;
  }
  el[idx] = s1;
  er[idx] = s2;
}

// ---- per-edge score + segment max over dst --------------------------------

__global__ void edge_score_kernel(const int* __restrict__ src,
                                  const int* __restrict__ dst,
                                  const float* __restrict__ el,
                                  const float* __restrict__ er,
                                  float* __restrict__ ebuf,
                                  float* __restrict__ emax, int E) {
  int e = blockIdx.x * blockDim.x + threadIdx.x;
  if (e >= E) return;
  int s = src[e], d = dst[e];
  float4 lv = *(const float4*)(el + 4 * (size_t)s);
  float4 rv = *(const float4*)(er + 4 * (size_t)d);
  float sc[4] = {lv.x + rv.x, lv.y + rv.y, lv.z + rv.z, lv.w + rv.w};
  float4 o;
  float* op = &o.x;
#pragma unroll
  for (int h = 0; h < 4; ++h) {
    float x = sc[h];
    x = x > 0.f ? x : SLOPE * x;
    op[h] = x;
    atomicMaxF(emax + 4 * (size_t)d + h, x);
  }
  *(float4*)(ebuf + 4 * (size_t)e) = o;
}

// ---- exp(score - max) + segment sum over dst ------------------------------

__global__ void edge_exp_kernel(const int* __restrict__ dst,
                                float* __restrict__ ebuf,
                                const float* __restrict__ emax,
                                float* __restrict__ denom, int E) {
  int e = blockIdx.x * blockDim.x + threadIdx.x;
  if (e >= E) return;
  int d = dst[e];
  float4 sv = *(const float4*)(ebuf + 4 * (size_t)e);
  float4 mv = *(const float4*)(emax + 4 * (size_t)d);
  float ex[4] = {__expf(sv.x - mv.x), __expf(sv.y - mv.y),
                 __expf(sv.z - mv.z), __expf(sv.w - mv.w)};
  float4 o = {ex[0], ex[1], ex[2], ex[3]};
  *(float4*)(ebuf + 4 * (size_t)e) = o;
#pragma unroll
  for (int h = 0; h < 4; ++h) atomicAdd(denom + 4 * (size_t)d + h, ex[h]);
}

// ---- weighted message scatter-add: out[dst] += a * fs[src] ----------------
// one thread per (edge, float4 feature chunk)

__global__ void scatter_kernel(const int* __restrict__ src,
                               const int* __restrict__ dst,
                               const float* __restrict__ ebuf,
                               const float* __restrict__ denom,
                               const float* __restrict__ fs,
                               float* __restrict__ out, int E, int HD, int D) {
  long long idx = (long long)blockIdx.x * blockDim.x + threadIdx.x;
  int cpe = HD >> 2;                                  // float4 chunks / edge
  long long total = (long long)E * cpe;
  if (idx >= total) return;
  int e  = (int)(idx / cpe);
  int c  = (int)(idx % cpe);
  int fb = c * 4;
  int h  = fb / D;
  int s  = src[e], d = dst[e];
  float a = ebuf[4 * (size_t)e + h] / denom[4 * (size_t)d + h];
  float4 v = *(const float4*)(fs + (size_t)s * HD + fb);
  float* o = out + (size_t)d * HD + fb;
  atomicAdd(o + 0, v.x * a);
  atomicAdd(o + 1, v.y * a);
  atomicAdd(o + 2, v.z * a);
  atomicAdd(o + 3, v.w * a);
}

// ---- layer-1 epilogue: h1 = elu(gat + h), stored as bf16 ------------------

__global__ void finalize1_kernel(const float* __restrict__ gat,
                                 const float* __restrict__ h0,
                                 __bf16* __restrict__ h1bf, long long n) {
  long long i = (long long)blockIdx.x * blockDim.x + threadIdx.x;
  if (i >= n) return;
  float x = gat[i] + h0[i];
  x = x > 0.f ? x : (__expf(x) - 1.0f);
  h1bf[i] = f2bf(x);
}

// ---- layer-2 epilogue: out[n,c] = mean_h (gat2 + res2) --------------------

__global__ void final_mean_kernel(const float* __restrict__ gat2,
                                  const float* __restrict__ res2,
                                  float* __restrict__ out, int n_nodes) {
  int idx = blockIdx.x * blockDim.x + threadIdx.x;
  if (idx >= n_nodes * 32) return;
  int n = idx >> 5, c = idx & 31;
  size_t base = (size_t)n * 128 + c;
  float s = 0.f;
#pragma unroll
  for (int h = 0; h < 4; ++h) s += gat2[base + 32 * h] + res2[base + 32 * h];
  out[idx] = 0.25f * s;
}

// ---------------------------------------------------------------------------

extern "C" void kernel_launch(void* const* d_in, const int* in_sizes, int n_in,
                              void* d_out, int out_size, void* d_ws,
                              size_t ws_size, hipStream_t stream) {
  const float* h    = (const float*)d_in[0];
  const int*   src  = (const int*)d_in[1];
  const int*   dst  = (const int*)d_in[2];
  const float* W1s  = (const float*)d_in[3];
  const float* W1d  = (const float*)d_in[4];
  const float* al1  = (const float*)d_in[5];
  const float* ar1  = (const float*)d_in[6];
  const float* W2s  = (const float*)d_in[7];
  const float* W2d  = (const float*)d_in[8];
  const float* al2  = (const float*)d_in[9];
  const float* ar2  = (const float*)d_in[10];
  const float* Wres = (const float*)d_in[11];
  float* out = (float*)d_out;
  const int E = in_sizes[1];

  // ---- workspace carving (region reuse; peak ~145 MB) ----
  char* p = (char*)d_ws;
  size_t o = 0;
  __bf16* hbf = (__bf16*)(p + o);  o += (size_t)N_NODES * 256 * 2;   // also h1bf
  __bf16* w1s_p = (__bf16*)(p + o); o += 256 * 256 * 2;
  __bf16* w1d_p = (__bf16*)(p + o); o += 256 * 256 * 2;
  __bf16* w2s_p = (__bf16*)(p + o); o += 256 * 128 * 2;
  __bf16* w2d_p = (__bf16*)(p + o); o += 256 * 128 * 2;
  __bf16* wr_p  = (__bf16*)(p + o); o += 256 * 128 * 2;
  size_t o_fs1 = o;                 o += (size_t)N_NODES * 256 * 4;  // fs1 | fs2+fd2
  size_t o_fd1 = o;                 o += (size_t)N_NODES * 256 * 4;  // fd1/gat1 | res2+gat2
  float* el    = (float*)(p + o);   o += (size_t)N_NODES * 4 * 4;
  float* er    = (float*)(p + o);   o += (size_t)N_NODES * 4 * 4;
  float* emax  = (float*)(p + o);   o += (size_t)N_NODES * 4 * 4;
  float* denom = (float*)(p + o);   o += (size_t)N_NODES * 4 * 4;
  float* ebuf  = (float*)(p + o);   o += (size_t)E * 4 * 4;

  float* fs1  = (float*)(p + o_fs1);
  float* fd1  = (float*)(p + o_fd1);
  float* gat1 = fd1;                                        // reuse after el/er
  __bf16* h1bf = hbf;                                       // reuse after gemms
  float* fs2  = (float*)(p + o_fs1);
  float* fd2  = (float*)(p + o_fs1 + (size_t)N_NODES * 128 * 4);
  float* res2 = (float*)(p + o_fd1);
  float* gat2 = (float*)(p + o_fd1 + (size_t)N_NODES * 128 * 4);

  const int BT = 256;
  auto grid = [](long long n, int bt) {
    return dim3((unsigned)((n + bt - 1) / bt));
  };
  auto gemm = [&](const __bf16* A, const __bf16* Bp, float* C, int Kin,
                  int Kout) {
    int mtiles  = (N_NODES + 15) / 16;
    int mgroups = (mtiles + 7) / 8;              // 8 m-tiles per block
    int blocks  = mgroups * (Kout / 64);
    gemm_bf16_wmma<<<blocks, 256, 0, stream>>>(A, Bp, C, N_NODES, Kin, Kout);
  };

  // ---- one-time converts / packs (cheap; rerun each call: deterministic) --
  long long nh = (long long)N_NODES * 256;
  cvt_f32_bf16<<<grid(nh, BT), BT, 0, stream>>>(h, hbf, nh);
  pack_w_bf16<<<grid(256 * 256, BT), BT, 0, stream>>>(W1s, w1s_p, 256, 256);
  pack_w_bf16<<<grid(256 * 256, BT), BT, 0, stream>>>(W1d, w1d_p, 256, 256);
  pack_w_bf16<<<grid(256 * 128, BT), BT, 0, stream>>>(W2s, w2s_p, 256, 128);
  pack_w_bf16<<<grid(256 * 128, BT), BT, 0, stream>>>(W2d, w2d_p, 256, 128);
  pack_w_bf16<<<grid(256 * 128, BT), BT, 0, stream>>>(Wres, wr_p, 256, 128);

  // ================= layer 1 =================
  gemm(hbf, w1s_p, fs1, 256, 256);
  gemm(hbf, w1d_p, fd1, 256, 256);
  el_er_kernel<<<grid((long long)N_NODES * 4, BT), BT, 0, stream>>>(
      fs1, fd1, al1, ar1, el, er, N_NODES, 64);
  fill_f32<<<grid(nh, BT), BT, 0, stream>>>(gat1, 0.f, nh);   // overwrites fd1
  fill_f32<<<grid((long long)N_NODES * 4, BT), BT, 0, stream>>>(
      emax, -INFINITY, (long long)N_NODES * 4);
  fill_f32<<<grid((long long)N_NODES * 4, BT), BT, 0, stream>>>(
      denom, 0.f, (long long)N_NODES * 4);
  edge_score_kernel<<<grid(E, BT), BT, 0, stream>>>(src, dst, el, er, ebuf,
                                                    emax, E);
  edge_exp_kernel<<<grid(E, BT), BT, 0, stream>>>(dst, ebuf, emax, denom, E);
  scatter_kernel<<<grid((long long)E * 64, BT), BT, 0, stream>>>(
      src, dst, ebuf, denom, fs1, gat1, E, 256, 64);
  finalize1_kernel<<<grid(nh, BT), BT, 0, stream>>>(gat1, h, h1bf, nh);

  // ================= layer 2 =================
  gemm(h1bf, w2s_p, fs2, 256, 128);
  gemm(h1bf, w2d_p, fd2, 256, 128);
  gemm(h1bf, wr_p, res2, 256, 128);                         // overwrites gat1
  el_er_kernel<<<grid((long long)N_NODES * 4, BT), BT, 0, stream>>>(
      fs2, fd2, al2, ar2, el, er, N_NODES, 32);
  long long n2 = (long long)N_NODES * 128;
  fill_f32<<<grid(n2, BT), BT, 0, stream>>>(gat2, 0.f, n2);
  fill_f32<<<grid((long long)N_NODES * 4, BT), BT, 0, stream>>>(
      emax, -INFINITY, (long long)N_NODES * 4);
  fill_f32<<<grid((long long)N_NODES * 4, BT), BT, 0, stream>>>(
      denom, 0.f, (long long)N_NODES * 4);
  edge_score_kernel<<<grid(E, BT), BT, 0, stream>>>(src, dst, el, er, ebuf,
                                                    emax, E);
  edge_exp_kernel<<<grid(E, BT), BT, 0, stream>>>(dst, ebuf, emax, denom, E);
  scatter_kernel<<<grid((long long)E * 32, BT), BT, 0, stream>>>(
      src, dst, ebuf, denom, fs2, gat2, E, 128, 32);
  final_mean_kernel<<<grid((long long)N_NODES * 32, BT), BT, 0, stream>>>(
      gat2, res2, out, N_NODES);
}
